// MoEDecoderLayer_31181462569069
// MI455X (gfx1250) — matrix operand order repaired and checked by reading, not compiled, not run on hardware
//
#include <hip/hip_runtime.h>
#include <hip/hip_bf16.h>
#include <cmath>

// ---------------------------------------------------------------------------
// Problem constants (match the reference)
// ---------------------------------------------------------------------------
constexpr int Bc   = 2;
constexpr int Sc   = 2048;
constexpr int Hc   = 1024;
constexpr int NHc  = 16;
constexpr int HDc  = 64;
constexpr int Ec   = 32;     // experts
constexpr int KTOP = 4;      // top-k
constexpr int Ic   = 512;    // moe intermediate
constexpr int ISHc = 2048;   // shared-expert intermediate
constexpr int CAPc = 1024;   // per-expert capacity
constexpr int Tc   = Bc * Sc;        // 4096 tokens
constexpr int Rc   = Tc * KTOP;      // 16384 routed pairs

typedef __bf16 bf16_t;
typedef bf16_t v16bf __attribute__((ext_vector_type(16)));
typedef bf16_t v8bf  __attribute__((ext_vector_type(8)));
typedef bf16_t v4bf  __attribute__((ext_vector_type(4)));
typedef float  v8f   __attribute__((ext_vector_type(8)));
typedef float  f32x4 __attribute__((ext_vector_type(4)));

__device__ __forceinline__ v16bf mk16(v8bf lo, v8bf hi) {
    v16bf r;
#pragma unroll
    for (int i = 0; i < 8; ++i) { r[i] = lo[i]; r[8 + i] = hi[i]; }
    return r;
}

__device__ __forceinline__ v8f wmma_bf16(v16bf a, v16bf b, v8f c) {
    // D = A(16x32) * B(32x16) + C, fp32 accumulate (gfx1250 wave32 WMMA)
    return __builtin_amdgcn_wmma_f32_16x16x32_bf16(false, a, false, b,
                                                   (short)0, c, false, false);
}

// ---------------------------------------------------------------------------
// RMSNorm: one token per block
// ---------------------------------------------------------------------------
__global__ __launch_bounds__(256) void k_rmsnorm(const float* __restrict__ x,
                                                 const float* __restrict__ w,
                                                 float* __restrict__ y) {
    __shared__ float red[256];
    const long long t = blockIdx.x;
    const float* xr = x + t * Hc;
    float ss = 0.f;
    for (int c = threadIdx.x; c < Hc; c += 256) { float v = xr[c]; ss += v * v; }
    red[threadIdx.x] = ss;
    __syncthreads();
    for (int o = 128; o > 0; o >>= 1) {
        if (threadIdx.x < o) red[threadIdx.x] += red[threadIdx.x + o];
        __syncthreads();
    }
    const float rs = rsqrtf(red[0] / (float)Hc + 1e-6f);
    for (int c = threadIdx.x; c < Hc; c += 256) y[t * Hc + c] = xr[c] * rs * w[c];
}

// ---------------------------------------------------------------------------
// Generic batched WMMA GEMM:  C[M,N] = A[M,K] * B[N,K]^T  (+ optional Add)
// fp32 in global, bf16 staged in LDS, fp32 accumulate.
// Block tile 128x128, K-tile 64, 8 waves (2x4), wave tile 64x32.
// Requires M%128==0, N%128==0, K%64==0 (true for every call site).
// ---------------------------------------------------------------------------
constexpr int GBM = 128, GBN = 128, GBK = 64, GLD = GBK + 8; // 72 elem stride

__global__ __launch_bounds__(256) void k_gemm(const float* __restrict__ A, long long sA,
                                              const float* __restrict__ Bw, long long sB,
                                              float* __restrict__ C, long long sC,
                                              const float* __restrict__ Add,
                                              int M, int N, int Kd) {
    __shared__ bf16_t As[GBM * GLD];
    __shared__ bf16_t Bs[GBN * GLD];

    const long long bz = blockIdx.z;
    A += bz * sA; Bw += bz * sB; C += bz * sC;
    if (Add) Add += bz * sC;

    const int m0 = blockIdx.y * GBM;
    const int n0 = blockIdx.x * GBN;
    const int t    = threadIdx.x;
    const int lane = t & 31;
    const int wid  = t >> 5;
    const int wm   = (wid & 1) * 64;   // wave M offset
    const int wn   = (wid >> 1) * 32;  // wave N offset

    v8f acc[4][2];
#pragma unroll
    for (int mt = 0; mt < 4; ++mt)
#pragma unroll
        for (int nt = 0; nt < 2; ++nt)
#pragma unroll
            for (int v = 0; v < 8; ++v) acc[mt][nt][v] = 0.f;

    for (int k0 = 0; k0 < Kd; k0 += GBK) {
        __syncthreads();
        // stage 128x64 tiles of A and B, fp32 -> bf16
#pragma unroll
        for (int i = 0; i < 8; ++i) {
            const int idx = t + i * 256;       // 0..2047
            const int row = idx >> 4;          // 0..127
            const int c4  = (idx & 15) << 2;   // 0..60 step 4
            f32x4 da = *(const f32x4*)(A + (long long)(m0 + row) * Kd + k0 + c4);
            f32x4 db = *(const f32x4*)(Bw + (long long)(n0 + row) * Kd + k0 + c4);
            v4bf ha, hb;
#pragma unroll
            for (int j = 0; j < 4; ++j) { ha[j] = (bf16_t)da[j]; hb[j] = (bf16_t)db[j]; }
            *(v4bf*)(As + row * GLD + c4) = ha;
            *(v4bf*)(Bs + row * GLD + c4) = hb;
            if (k0 + GBK < Kd) {
                __builtin_prefetch(A + (long long)(m0 + row) * Kd + k0 + GBK + c4, 0, 0);
                __builtin_prefetch(Bw + (long long)(n0 + row) * Kd + k0 + GBK + c4, 0, 0);
            }
        }
        __syncthreads();

        const int arow  = lane & 15;
        const int ahalf = (lane & 16) ? 8 : 0;   // A: K sub-offset per lane half
        const int bofs  = (lane & 16) ? 8 : 0;   // B: K sub-offset per lane half
#pragma unroll
        for (int kk = 0; kk < GBK; kk += 32) {
            v16bf af[4], bfg[2];
#pragma unroll
            for (int mt = 0; mt < 4; ++mt) {
                const bf16_t* p = As + (wm + mt * 16 + arow) * GLD + kk + ahalf;
                af[mt] = mk16(*(const v8bf*)p, *(const v8bf*)(p + 16));
            }
#pragma unroll
            for (int nt = 0; nt < 2; ++nt) {
                const bf16_t* p = Bs + (wn + nt * 16 + arow) * GLD + kk + bofs;
                bfg[nt] = mk16(*(const v8bf*)p, *(const v8bf*)(p + 16));
            }
#pragma unroll
            for (int mt = 0; mt < 4; ++mt)
#pragma unroll
                for (int nt = 0; nt < 2; ++nt)
                    acc[mt][nt] = wmma_bf16(af[mt], bfg[nt], acc[mt][nt]);
        }
    }

    // epilogue: C/D layout — row = v + 8*(lane>=16), col = lane&15
    const int cl = lane & 15;
    const int rh = (lane >> 4) << 3;
#pragma unroll
    for (int mt = 0; mt < 4; ++mt)
#pragma unroll
        for (int nt = 0; nt < 2; ++nt) {
            const int col = n0 + wn + nt * 16 + cl;
#pragma unroll
            for (int v = 0; v < 8; ++v) {
                const int row = m0 + wm + mt * 16 + v + rh;
                const long long o = (long long)row * N + col;
                float val = acc[mt][nt][v];
                if (Add) val += Add[o];
                C[o] = val;
            }
        }
}

// ---------------------------------------------------------------------------
// RoPE + bf16 convert for q,k (q pre-scaled by 1/sqrt(HD))
// layout: [B,S,NH,HD] flattened == [T, H]
// ---------------------------------------------------------------------------
__global__ __launch_bounds__(256) void k_rope(const float* __restrict__ qf,
                                              const float* __restrict__ kf,
                                              const float* __restrict__ cosb,
                                              const float* __restrict__ sinb,
                                              bf16_t* __restrict__ q16,
                                              bf16_t* __restrict__ k16) {
    const long long i = (long long)blockIdx.x * 256 + threadIdx.x; // over T*H
    const int d = (int)(i & 63);
    const long long tok = i >> 10;             // H = 1024
    const int s = (int)(tok & (Sc - 1));
    const float c  = cosb[s * HDc + d];
    const float sn = sinb[s * HDc + d];
    const float q = qf[i], k = kf[i];
    float qo, ko;
    if (d < 32) { qo = q * c - qf[i + 32] * sn; ko = k * c - kf[i + 32] * sn; }
    else        { qo = q * c + qf[i - 32] * sn; ko = k * c + kf[i - 32] * sn; }
    q16[i] = (bf16_t)(qo * 0.125f);            // 1/sqrt(64)
    k16[i] = (bf16_t)ko;
}

// V transpose+convert: vt[((b*NH+h)*HD+d)*S + s] = v[((b*S+s)*NH+h)*HD + d]
__global__ __launch_bounds__(256) void k_vtrans(const float* __restrict__ vf,
                                                bf16_t* __restrict__ vt) {
    const long long i = (long long)blockIdx.x * 256 + threadIdx.x; // over B*NH*HD*S
    const int s = (int)(i & (Sc - 1));
    long long r = i >> 11;
    const int d = (int)(r & 63); r >>= 6;
    const int h = (int)(r & 15); r >>= 4;
    const int b = (int)r;
    vt[i] = (bf16_t)vf[((long long)(b * Sc + s) * NHc + h) * HDc + d];
}

// ---------------------------------------------------------------------------
// Flash attention: one wave per (16 q rows, head, batch). Online softmax.
// Q pre-scaled; K/V fragments loaded directly from global in WMMA B layout.
// ---------------------------------------------------------------------------
__global__ __launch_bounds__(32) void k_flash(const bf16_t* __restrict__ Q,
                                              const bf16_t* __restrict__ Kb,
                                              const bf16_t* __restrict__ VT,
                                              float* __restrict__ O) {
    __shared__ bf16_t Pst[16 * 40];            // P tile, padded stride 40
    const int lane = threadIdx.x;
    const int q0 = blockIdx.x * 16;
    const int h  = blockIdx.y;
    const int b  = blockIdx.z;

    const int arow  = lane & 15;
    const int ahalf = (lane & 16) ? 8 : 0;
    const int rh    = (lane >> 4) << 3;

    // preload Q fragments (A layout), fixed for the whole key loop
    v16bf qf[2];
#pragma unroll
    for (int kk = 0; kk < 2; ++kk) {
        const bf16_t* p = Q + ((long long)(b * Sc + q0 + arow) * NHc + h) * HDc + kk * 32 + ahalf;
        qf[kk] = mk16(*(const v8bf*)p, *(const v8bf*)(p + 16));
    }

    v8f o[4];
    float m[8], l[8];
#pragma unroll
    for (int dt = 0; dt < 4; ++dt)
#pragma unroll
        for (int v = 0; v < 8; ++v) o[dt][v] = 0.f;
#pragma unroll
    for (int v = 0; v < 8; ++v) { m[v] = -3.0e38f; l[v] = 0.f; }

    for (int j0 = 0; j0 <= q0 + 15; j0 += 32) {
        // ---- scores: S = Q * K^T for 32 keys (two 16-col WMMA tiles) ----
        v8f s01[2];
#pragma unroll
        for (int nt = 0; nt < 2; ++nt) {
#pragma unroll
            for (int v = 0; v < 8; ++v) s01[nt][v] = 0.f;
            const int key  = j0 + nt * 16 + arow;
            const int dofs = (lane & 16) ? 8 : 0;
#pragma unroll
            for (int kk = 0; kk < 2; ++kk) {
                const bf16_t* p = Kb + ((long long)(b * Sc + key) * NHc + h) * HDc + kk * 32 + dofs;
                v16bf kfrag = mk16(*(const v8bf*)p, *(const v8bf*)(p + 16));
                s01[nt] = wmma_bf16(qf[kk], kfrag, s01[nt]);
            }
            // causal mask: this lane holds column key = j0+nt*16+(lane&15)
#pragma unroll
            for (int v = 0; v < 8; ++v) {
                const int qrow = q0 + v + rh;
                if (j0 + nt * 16 + arow > qrow) s01[nt][v] = -1.0e30f;
            }
        }
        // ---- online softmax (row reduce over 16 lanes of each half) ----
        float sc[8];
#pragma unroll
        for (int v = 0; v < 8; ++v) {
            float mx = fmaxf(s01[0][v], s01[1][v]);
            for (int off = 8; off > 0; off >>= 1) mx = fmaxf(mx, __shfl_xor(mx, off, 32));
            const float mnew = fmaxf(m[v], mx);
            const float p0 = __expf(s01[0][v] - mnew);
            const float p1 = __expf(s01[1][v] - mnew);
            float rs = p0 + p1;
            for (int off = 8; off > 0; off >>= 1) rs += __shfl_xor(rs, off, 32);
            const float scale = __expf(m[v] - mnew);
            l[v] = l[v] * scale + rs;
            m[v] = mnew;
            sc[v] = scale;
            Pst[(v + rh) * 40 + arow]      = (bf16_t)p0;
            Pst[(v + rh) * 40 + 16 + arow] = (bf16_t)p1;
        }
#pragma unroll
        for (int dt = 0; dt < 4; ++dt)
#pragma unroll
            for (int v = 0; v < 8; ++v) o[dt][v] *= sc[v];

        // wave-local LDS ordering (CDNA5 split counter)
        asm volatile("s_wait_dscnt 0" ::: "memory");

        // ---- O += P * V  (contraction over 32 keys) ----
        const bf16_t* pp = Pst + arow * 40 + ahalf;
        v16bf pa = mk16(*(const v8bf*)pp, *(const v8bf*)(pp + 16));
        const int kofs = (lane & 16) ? 8 : 0;
#pragma unroll
        for (int dt = 0; dt < 4; ++dt) {
            const int dim = dt * 16 + arow;
            const bf16_t* vp = VT + ((long long)(b * NHc + h) * HDc + dim) * Sc + j0 + kofs;
            v16bf vfrag = mk16(*(const v8bf*)vp, *(const v8bf*)(vp + 16));
            o[dt] = wmma_bf16(pa, vfrag, o[dt]);
        }
    }

    // write out: ao[(b*S+s)*H + h*64 + dim]
#pragma unroll
    for (int dt = 0; dt < 4; ++dt)
#pragma unroll
        for (int v = 0; v < 8; ++v) {
            const int row = q0 + v + rh;
            O[(long long)(b * Sc + row) * Hc + h * 64 + dt * 16 + arow] = o[dt][v] / l[v];
        }
}

// ---------------------------------------------------------------------------
// Router: one wave per token; lane == expert (E==32==wave32).
// Emits raw logits (to d_out tail) and renormalized top-4 weights/ids.
// ---------------------------------------------------------------------------
__global__ __launch_bounds__(32) void k_router(const float* __restrict__ h2,
                                               const float* __restrict__ gw,
                                               float* __restrict__ logits,
                                               float* __restrict__ tw,
                                               int* __restrict__ tidx) {
    __shared__ float xs[Hc];
    const int t = blockIdx.x;
    const int lane = threadIdx.x;
    for (int c = lane; c < Hc; c += 32) xs[c] = h2[(long long)t * Hc + c];
    asm volatile("s_wait_dscnt 0" ::: "memory");

    float acc = 0.f;
    const float* g = gw + (long long)lane * Hc;
    for (int c = 0; c < Hc; ++c) acc += xs[c] * g[c];
    logits[(long long)t * Ec + lane] = acc;

    float mx = acc;
    for (int off = 16; off > 0; off >>= 1) mx = fmaxf(mx, __shfl_xor(mx, off, 32));
    float p = __expf(acc - mx);
    float sum = p;
    for (int off = 16; off > 0; off >>= 1) sum += __shfl_xor(sum, off, 32);
    p /= sum;

    float myp = p, tsum = 0.f;
    float selp[KTOP]; int seli[KTOP];
#pragma unroll
    for (int k = 0; k < KTOP; ++k) {
        float v = myp; int idx = lane;
        for (int off = 16; off > 0; off >>= 1) {
            float vo = __shfl_xor(v, off, 32);
            int   io = __shfl_xor(idx, off, 32);
            if (vo > v || (vo == v && io < idx)) { v = vo; idx = io; }
        }
        selp[k] = v; seli[k] = idx; tsum += v;
        if (lane == idx) myp = -1.f;
    }
    if (lane < KTOP) {
        tw[(long long)t * KTOP + lane]  = selp[lane] / tsum;
        tidx[t * KTOP + lane]           = seli[lane];
    }
}

__global__ void k_zero_counts(int* cnt) { if (threadIdx.x < Ec) cnt[threadIdx.x] = 0; }

__global__ __launch_bounds__(256) void k_assign(const int* __restrict__ tidx,
                                                int* __restrict__ cnt,
                                                int* __restrict__ dest) {
    const int r = blockIdx.x * 256 + threadIdx.x;
    if (r >= Rc) return;
    const int e = tidx[r];
    const int pos = atomicAdd(&cnt[e], 1);
    dest[r] = (pos < CAPc) ? e * CAPc + pos : -1;   // -1 == dropped
}

__global__ __launch_bounds__(256) void k_scatter(const float* __restrict__ h2,
                                                 const int* __restrict__ dest,
                                                 float* __restrict__ xb) {
    const int r = blockIdx.x;
    const int d = dest[r];
    if (d < 0) return;
    const float* src = h2 + (long long)(r >> 2) * Hc;  // token = r / KTOP
    float* dst = xb + (long long)d * Hc;
    for (int c = threadIdx.x; c < Hc; c += 256) dst[c] = src[c];
}

// SwiGLU: out[row, f] = silu(G[row*stride+f]) * U[row*stride+f]
__global__ __launch_bounds__(256) void k_swiglu(const float* __restrict__ G,
                                                const float* __restrict__ U,
                                                float* __restrict__ out,
                                                long long total, int inner, int instride) {
    const long long i = (long long)blockIdx.x * 256 + threadIdx.x;
    if (i >= total) return;
    const long long row = i / inner;
    const int f = (int)(i % inner);
    const float g = G[row * instride + f];
    const float u = U[row * instride + f];
    out[i] = g / (1.f + __expf(-g)) * u;
}

__global__ __launch_bounds__(256) void k_combine(const float* __restrict__ hres,
                                                 const float* __restrict__ yb,
                                                 const int* __restrict__ dest,
                                                 const float* __restrict__ tw,
                                                 const float* __restrict__ shexp,
                                                 float* __restrict__ out) {
    const int t = blockIdx.x;
    for (int c = threadIdx.x; c < Hc; c += 256) {
        float acc = hres[(long long)t * Hc + c] + shexp[(long long)t * Hc + c];
#pragma unroll
        for (int k = 0; k < KTOP; ++k) {
            const int d = dest[t * KTOP + k];
            if (d >= 0) acc += tw[t * KTOP + k] * yb[(long long)d * Hc + c];
        }
        out[(long long)t * Hc + c] = acc;
    }
}

// ---------------------------------------------------------------------------
// Host-side orchestration
// ---------------------------------------------------------------------------
extern "C" void kernel_launch(void* const* d_in, const int* in_sizes, int n_in,
                              void* d_out, int out_size, void* d_ws, size_t ws_size,
                              hipStream_t stream) {
    const float* x        = (const float*)d_in[0];
    const float* cosb     = (const float*)d_in[1];
    const float* sinb     = (const float*)d_in[2];
    const float* wq       = (const float*)d_in[3];
    const float* wk       = (const float*)d_in[4];
    const float* wv       = (const float*)d_in[5];
    const float* wo       = (const float*)d_in[6];
    const float* norm1_w  = (const float*)d_in[7];
    const float* norm2_w  = (const float*)d_in[8];
    const float* gate_w   = (const float*)d_in[9];
    const float* w13      = (const float*)d_in[10];
    const float* w2       = (const float*)d_in[11];
    const float* sh_gate  = (const float*)d_in[12];
    const float* sh_up    = (const float*)d_in[13];
    const float* sh_down  = (const float*)d_in[14];

    float* out    = (float*)d_out;
    float* logits = out + (size_t)Tc * Hc;

    // ---- workspace carve-out ----
    char* ws = (char*)d_ws;
    size_t off = 0;
    auto alloc = [&](size_t bytes) { size_t r = off; off += (bytes + 255) & ~(size_t)255; return r; };
    const size_t TH4 = (size_t)Tc * Hc * 4, TH2 = (size_t)Tc * Hc * 2;
    float*  h1   = (float*)(ws + alloc(TH4));
    float*  qf   = (float*)(ws + alloc(TH4));
    float*  kf   = (float*)(ws + alloc(TH4));
    float*  vf   = (float*)(ws + alloc(TH4));
    bf16_t* q16  = (bf16_t*)(ws + alloc(TH2));
    bf16_t* k16  = (bf16_t*)(ws + alloc(TH2));
    bf16_t* vt16 = (bf16_t*)(ws + alloc(TH2));
    float*  hres = (float*)(ws + alloc(TH4));
    float*  h2   = (float*)(ws + alloc(TH4));
    float*  tw   = (float*)(ws + alloc((size_t)Rc * 4));
    int*    tidx = (int*)(ws + alloc((size_t)Rc * 4));
    int*    dest = (int*)(ws + alloc((size_t)Rc * 4));
    int*    cnt  = (int*)(ws + alloc((size_t)Ec * 4));
    float*  xb   = (float*)(ws + alloc((size_t)Ec * CAPc * Hc * 4));      // 128 MB
    float*  gu   = (float*)(ws + alloc((size_t)Ec * CAPc * 2 * Ic * 4));  // 128 MB
    float*  act  = (float*)(ws + alloc((size_t)Ec * CAPc * Ic * 4));      //  64 MB
    float*  yb   = (float*)(ws + alloc((size_t)Ec * CAPc * Hc * 4));      // 128 MB
    // dead-buffer reuse: qf region holds ao after RoPE consumed it;
    // xb region holds shared-expert activations after the grouped GEMM consumed it.
    float* ao     = qf;
    float* sg     = xb;                                          // T*ISH
    float* su     = (float*)((char*)xb + (size_t)Tc * ISHc * 4);
    float* sact   = (float*)((char*)xb + (size_t)2 * Tc * ISHc * 4);
    float* shexp  = (float*)((char*)xb + (size_t)3 * Tc * ISHc * 4);      // T*H
    (void)n_in; (void)in_sizes; (void)out_size; (void)ws_size;

    // 1) pre-attention RMSNorm
    k_rmsnorm<<<Tc, 256, 0, stream>>>(x, norm1_w, h1);

    // 2) QKV projections (WMMA GEMM, weights are [O,H] row-major == B[N,K])
    dim3 gP(Hc / GBN, Tc / GBM, 1);
    k_gemm<<<gP, 256, 0, stream>>>(h1, 0, wq, 0, qf, 0, nullptr, Tc, Hc, Hc);
    k_gemm<<<gP, 256, 0, stream>>>(h1, 0, wk, 0, kf, 0, nullptr, Tc, Hc, Hc);
    k_gemm<<<gP, 256, 0, stream>>>(h1, 0, wv, 0, vf, 0, nullptr, Tc, Hc, Hc);

    // 3) RoPE + bf16 convert; V transpose for P*V fragments
    const int nTH = (Tc * Hc) / 256;
    k_rope<<<nTH, 256, 0, stream>>>(qf, kf, cosb, sinb, q16, k16);
    k_vtrans<<<nTH, 256, 0, stream>>>(vf, vt16);

    // 4) flash attention (one wave per 16 q rows)
    k_flash<<<dim3(Sc / 16, NHc, Bc), 32, 0, stream>>>(q16, k16, vt16, ao);

    // 5) output projection + residual: h = x + ao*wo^T
    k_gemm<<<gP, 256, 0, stream>>>(ao, 0, wo, 0, hres, 0, x, Tc, Hc, Hc);

    // 6) post-attention RMSNorm
    k_rmsnorm<<<Tc, 256, 0, stream>>>(hres, norm2_w, h2);

    // 7) router: logits + top-4
    k_router<<<Tc, 32, 0, stream>>>(h2, gate_w, logits, tw, tidx);

    // 8) dispatch
    k_zero_counts<<<1, 32, 0, stream>>>(cnt);
    k_assign<<<Rc / 256, 256, 0, stream>>>(tidx, cnt, dest);
    k_scatter<<<Rc, 256, 0, stream>>>(h2, dest, xb);

    // 9) grouped expert FFN: gu = xb * w13^T  (batch = E)
    k_gemm<<<dim3((2 * Ic) / GBN, CAPc / GBM, Ec), 256, 0, stream>>>(
        xb, (long long)CAPc * Hc, w13, (long long)2 * Ic * Hc,
        gu, (long long)CAPc * 2 * Ic, nullptr, CAPc, 2 * Ic, Hc);

    // 10) fused SwiGLU on gu
    {
        const long long total = (long long)Ec * CAPc * Ic;
        k_swiglu<<<(int)((total + 255) / 256), 256, 0, stream>>>(gu, gu + Ic, act, total, Ic, 2 * Ic);
    }

    // 11) yb = act * w2^T (batch = E)
    k_gemm<<<dim3(Hc / GBN, CAPc / GBM, Ec), 256, 0, stream>>>(
        act, (long long)CAPc * Ic, w2, (long long)Hc * Ic,
        yb, (long long)CAPc * Hc, nullptr, CAPc, Hc, Ic);

    // 12) shared experts (buffers live in the retired xb region)
    k_gemm<<<dim3(ISHc / GBN, Tc / GBM, 1), 256, 0, stream>>>(h2, 0, sh_gate, 0, sg, 0, nullptr, Tc, ISHc, Hc);
    k_gemm<<<dim3(ISHc / GBN, Tc / GBM, 1), 256, 0, stream>>>(h2, 0, sh_up,   0, su, 0, nullptr, Tc, ISHc, Hc);
    {
        const long long total = (long long)Tc * ISHc;
        k_swiglu<<<(int)((total + 255) / 256), 256, 0, stream>>>(sg, su, sact, total, ISHc, ISHc);
    }
    k_gemm<<<dim3(Hc / GBN, Tc / GBM, 1), 256, 0, stream>>>(sact, 0, sh_down, 0, shexp, 0, nullptr, Tc, Hc, ISHc);

    // 13) combine: out = residual + routed + shared
    k_combine<<<Tc, 256, 0, stream>>>(hres, yb, dest, tw, shexp, out);
}
